// ElmoLstm_36490042147344
// MI455X (gfx1250) — matrix operand order repaired and checked by reading, not compiled, not run on hardware
//
#include <hip/hip_runtime.h>

// ---------------------------------------------------------------------------
// ELMo bi-LSTM-with-projection for gfx1250 (MI455X).
// Per-timestep fused gate GEMM [32x1024]x[1024x16384] + projection GEMM
// [32x128]x[128x512] per workgroup slice, all via v_wmma_f32_16x16x32_bf16
// (bf16 operands, fp32 accumulate). Cell state lives in VGPRs across steps.
// Cross-WG reduction via global_atomic_add_f32 + persistent grid barriers.
// ---------------------------------------------------------------------------

typedef __bf16 bf16_t;
typedef bf16_t v16bf __attribute__((ext_vector_type(16)));
typedef float  v8f   __attribute__((ext_vector_type(8)));
typedef unsigned int v4u __attribute__((ext_vector_type(4)));

union Frag { v16bf v; v4u q[2]; };

#define NB  32      // batch
#define NT  128     // time
#define NI  512     // input / hidden
#define NC  4096    // cell
#define NG  16384   // 4*NC gate width
#define KK  1024    // fused K = input(512) + hidden(512)

__device__ __forceinline__ bf16_t f2bf(float f) {
  unsigned u = __builtin_bit_cast(unsigned, f);
  unsigned r = (u + 0x7FFFu + ((u >> 16) & 1u)) >> 16;
  unsigned short s = (unsigned short)r;
  return __builtin_bit_cast(bf16_t, s);
}
__device__ __forceinline__ float sigf(float x) { return 1.0f / (1.0f + __expf(-x)); }
__device__ __forceinline__ float clip3(float x) { return fminf(3.0f, fmaxf(-3.0f, x)); }

// Monotonic-counter grid barrier (all member WGs must be co-resident).
__device__ __forceinline__ void gbar(unsigned* cnt, unsigned target) {
  __syncthreads();
  if (threadIdx.x == 0) {
    __threadfence();
    atomicAdd(cnt, 1u);
    while (*((volatile unsigned*)cnt) < target) __builtin_amdgcn_s_sleep(1);
    __threadfence();
  }
  __syncthreads();
}

// ---------------------------------------------------------------------------
// Prep kernels: zero state, cast input to bf16, repack weights fragment-major.
// B-fragment layout per (ntile,kchunk): 512 bf16 = 32 lanes x 16 values.
// Lane l -> column n = ntile*16 + (l&15); K = kchunk*32 + ((l&16)?16:0) + e.
// ---------------------------------------------------------------------------

__global__ void prep_zero(float* h_acc, float* h_prev, bf16_t* h_b, unsigned* bars) {
  int i = threadIdx.x;
  if (i < 16) bars[i] = 0u;
  for (int k = i; k < 2 * NB * NI; k += 256) {
    h_acc[k] = 0.0f; h_prev[k] = 0.0f; h_b[k] = f2bf(0.0f);
  }
}

__global__ void prep_x(const float* __restrict__ x, bf16_t* __restrict__ xb, int n) {
  int i = blockIdx.x * 256 + threadIdx.x;
  if (i < n) xb[i] = f2bf(x[i]);
}

// Pack [Wx;Wh] -> w1: per layer-dir, 1024 ntiles x 32 kchunks x 512 bf16.
__global__ void prep_pack_w1(const float* __restrict__ Wx, const float* __restrict__ Wh,
                             bf16_t* __restrict__ w1) {
  size_t p = ((size_t)blockIdx.x * 256 + threadIdx.x) * 8;
  if (p >= (size_t)4 * NG * KK) return;
  int e0    = (int)(p & 15);          // 0 or 8
  int lane  = (int)((p >> 4) & 31);
  int kc    = (int)((p >> 9) & 31);
  int ntile = (int)((p >> 14) & 1023);
  int ld    = (int)(p >> 24);
  int n     = ntile * 16 + (lane & 15);
  int kbase = kc * 32 + ((lane & 16) ? 16 : 0) + e0;   // contiguous run of 8
  const float* src = (kbase < NI)
      ? (Wx + (((size_t)ld * NG + n) * NI + kbase))
      : (Wh + (((size_t)ld * NG + n) * NI + (kbase - NI)));
  bf16_t* dst = w1 + p;
#pragma unroll
  for (int i = 0; i < 8; ++i) dst[i] = f2bf(src[i]);
}

// Pack Wp -> wp: per layer-dir, 32 ntiles x 128 kchunks x 512 bf16.
__global__ void prep_pack_wp(const float* __restrict__ Wp, bf16_t* __restrict__ wp) {
  size_t p = ((size_t)blockIdx.x * 256 + threadIdx.x) * 8;
  if (p >= (size_t)4 * NC * NI) return;
  int e0   = (int)(p & 15);
  int lane = (int)((p >> 4) & 31);
  int kc   = (int)((p >> 9) & 127);
  int nt   = (int)((p >> 16) & 31);
  int ld   = (int)(p >> 21);
  int nj   = nt * 16 + (lane & 15);
  int cb   = kc * 32 + ((lane & 16) ? 16 : 0) + e0;
  const float* src = Wp + (((size_t)ld * NI + nj) * NC + cb);
  bf16_t* dst = wp + p;
#pragma unroll
  for (int i = 0; i < 8; ++i) dst[i] = f2bf(src[i]);
}

// ---------------------------------------------------------------------------
// Main persistent kernel: 64 WGs x 256 threads.
//   dir   = blockIdx.x >> 5 (0=fwd, 1=bwd), slice = blockIdx.x & 31.
//   Slice owns 128 cell columns; wave w owns 16 of them (gate ntiles
//   g*256 + slice*8 + w for g=i,f,m,o), so the cell update is wave-local.
// ---------------------------------------------------------------------------
__global__ __launch_bounds__(256) void lstm_main(
    const int*   __restrict__ mask,
    const float* __restrict__ bias,
    float*       __restrict__ out,
    const bf16_t* __restrict__ x0,
    bf16_t*      __restrict__ y0,
    const bf16_t* __restrict__ w1all,
    const bf16_t* __restrict__ wpall,
    float*       __restrict__ h_acc_all,
    float*       __restrict__ h_prev_all,
    bf16_t*      __restrict__ h_b_all,
    unsigned*    __restrict__ bars) {
  __shared__ bf16_t A_lds[NB * KK];      // [x_t ; h]  row-major, 64 KB
  __shared__ bf16_t u_lds[NB * 128];     // o*tanh(c) slice, 8 KB
  __shared__ float  mask_lds[NB];

  const int tid    = threadIdx.x;
  const int lane   = tid & 31;
  const int wv     = tid >> 5;           // wave id 0..7 (wave32)
  const int dir    = blockIdx.x >> 5;
  const int slice  = blockIdx.x & 31;
  const int laneLo = lane & 15;
  const int laneHi = lane >> 4;
  const int kb     = (lane & 16) ? 16 : 0;  // A-fragment lane-half K offset (contiguous)

  float*    h_acc  = h_acc_all  + dir * (NB * NI);
  float*    h_prev = h_prev_all + dir * (NB * NI);
  bf16_t*   h_b    = h_b_all    + dir * (NB * NI);
  unsigned* bar_d  = bars + dir;
  unsigned* bar_a  = bars + 2;
  unsigned  ep_d = 0, ep_a = 0;

  const v8f z8 = {0.f, 0.f, 0.f, 0.f, 0.f, 0.f, 0.f, 0.f};

  for (int layer = 0; layer < 2; ++layer) {
    const int ld = layer * 2 + dir;
    const bf16_t* w1 = w1all + (size_t)ld * ((size_t)NG * KK);
    const bf16_t* wp = wpall + (size_t)ld * ((size_t)NC * NI);
    const float*  bv = bias + (size_t)ld * NG;
    const bf16_t* xsrc = (layer == 0) ? x0 : (y0 + (size_t)dir * NB * NT * NI);

    float creg[2][8];
#pragma unroll
    for (int mt = 0; mt < 2; ++mt)
#pragma unroll
      for (int e = 0; e < 8; ++e) creg[mt][e] = 0.0f;

    for (int step = 0; step < NT; ++step) {
      const int tt = dir ? (NT - 1 - step) : step;

      // ---- stage mask + A = [x_t ; h_{t-1}] into LDS (16B chunks) ----
      if (tid < NB) mask_lds[tid] = (float)mask[tid * NT + tt];
      for (int ch = tid; ch < NB * 128; ch += 256) {
        int row = ch >> 7, c = ch & 127;
        v4u val;
        if (c < 64)
          val = ((const v4u*)(xsrc + ((size_t)row * NT + tt) * NI))[c];
        else
          val = ((const v4u*)(h_b + row * NI))[c - 64];
        ((v4u*)A_lds)[ch] = val;
      }
      __syncthreads();

      // ---- GEMM1: gates[32 x 4*16] = A[32x1024] x W1[1024 x .] ----
      v8f acc[2][4];
#pragma unroll
      for (int mt = 0; mt < 2; ++mt)
#pragma unroll
        for (int g = 0; g < 4; ++g) acc[mt][g] = z8;

      const bf16_t* Ar0 = A_lds + laneLo * KK;
      const bf16_t* Ar1 = A_lds + (laneLo + 16) * KK;
      for (int kc = 0; kc < KK; kc += 32) {
        Frag a0, a1;
        a0.q[0] = *(const v4u*)(Ar0 + kc + kb);
        a0.q[1] = *(const v4u*)(Ar0 + kc + kb + 8);
        a1.q[0] = *(const v4u*)(Ar1 + kc + kb);
        a1.q[1] = *(const v4u*)(Ar1 + kc + kb + 8);
#pragma unroll
        for (int g = 0; g < 4; ++g) {
          const bf16_t* bp = w1 +
              ((size_t)((g * 256 + slice * 8 + wv) * 32 + (kc >> 5))) * 512 + lane * 16;
          Frag bb;
          bb.q[0] = ((const v4u*)bp)[0];
          bb.q[1] = ((const v4u*)bp)[1];
          acc[0][g] = __builtin_amdgcn_wmma_f32_16x16x32_bf16(
              false, a0.v, false, bb.v, (short)0, acc[0][g], false, false);
          acc[1][g] = __builtin_amdgcn_wmma_f32_16x16x32_bf16(
              false, a1.v, false, bb.v, (short)0, acc[1][g], false, false);
        }
      }

      // ---- bias + cell update (wave-local; C/D layouts align elementwise) ----
      {
        const int colbase = slice * 128 + wv * 16 + laneLo;   // cell column
        const float bi = bv[0 * NC + colbase];
        const float bf_ = bv[1 * NC + colbase];
        const float bm = bv[2 * NC + colbase];
        const float bo = bv[3 * NC + colbase];
#pragma unroll
        for (int mt = 0; mt < 2; ++mt) {
#pragma unroll
          for (int e = 0; e < 8; ++e) {
            int row = e + laneHi * 8 + mt * 16;               // batch index
            float mval = mask_lds[row];
            float gi = sigf(acc[mt][0][e] + bi);
            float gf = sigf(acc[mt][1][e] + bf_);
            float gm = tanhf(acc[mt][2][e] + bm);
            float go = sigf(acc[mt][3][e] + bo);
            float cn = clip3(gi * gm + gf * creg[mt][e]);
            float u  = go * tanhf(cn);
            creg[mt][e] = mval * cn + (1.0f - mval) * creg[mt][e];
            u_lds[row * 128 + wv * 16 + laneLo] = f2bf(u);
          }
        }
      }
      __syncthreads();

      // ---- GEMM2: partial h_new[32x512] = u[32x128] x Wp[128x512] ----
      v8f acc2[2][4];
#pragma unroll
      for (int mt = 0; mt < 2; ++mt)
#pragma unroll
        for (int q = 0; q < 4; ++q) acc2[mt][q] = z8;

      const bf16_t* Ur0 = u_lds + laneLo * 128;
      const bf16_t* Ur1 = u_lds + (laneLo + 16) * 128;
#pragma unroll
      for (int kc2 = 0; kc2 < 4; ++kc2) {
        Frag a0, a1;
        a0.q[0] = *(const v4u*)(Ur0 + kc2 * 32 + kb);
        a0.q[1] = *(const v4u*)(Ur0 + kc2 * 32 + kb + 8);
        a1.q[0] = *(const v4u*)(Ur1 + kc2 * 32 + kb);
        a1.q[1] = *(const v4u*)(Ur1 + kc2 * 32 + kb + 8);
#pragma unroll
        for (int q = 0; q < 4; ++q) {
          const bf16_t* bp = wp +
              ((size_t)((wv * 4 + q) * 128 + slice * 4 + kc2)) * 512 + lane * 16;
          Frag bb;
          bb.q[0] = ((const v4u*)bp)[0];
          bb.q[1] = ((const v4u*)bp)[1];
          acc2[0][q] = __builtin_amdgcn_wmma_f32_16x16x32_bf16(
              false, a0.v, false, bb.v, (short)0, acc2[0][q], false, false);
          acc2[1][q] = __builtin_amdgcn_wmma_f32_16x16x32_bf16(
              false, a1.v, false, bb.v, (short)0, acc2[1][q], false, false);
        }
      }

      // ---- cross-WG reduction of the K=4096 projection via f32 atomics ----
#pragma unroll
      for (int mt = 0; mt < 2; ++mt)
#pragma unroll
        for (int q = 0; q < 4; ++q)
#pragma unroll
          for (int e = 0; e < 8; ++e) {
            int row = e + laneHi * 8 + mt * 16;
            int col = (wv * 4 + q) * 16 + laneLo;
            atomicAdd(h_acc + row * NI + col, acc2[mt][q][e]);
          }
      __threadfence();

      ++ep_d; gbar(bar_d, 32u * ep_d);

      // ---- finalize: clip, mask-blend h, emit outputs, re-zero accum ----
      if (slice == 0) {
        for (int i = tid; i < NB * NI; i += 256) {
          int bb_ = i >> 9, j = i & 511;
          float hv = clip3(h_acc[i]);
          float mval = mask_lds[bb_];
          float h2 = mval * hv + (1.0f - mval) * h_prev[i];
          float y  = mval * hv;
          h_prev[i] = h2;
          h_b[i]    = f2bf(h2);
          h_acc[i]  = 0.0f;
          size_t oidx = (((size_t)layer * NB + bb_) * NT + tt) * (2 * NI) + dir * NI + j;
          if (layer == 0) {
            out[oidx] = y;
            y0[(size_t)dir * NB * NT * NI + ((size_t)bb_ * NT + tt) * NI + j] = f2bf(y);
          } else {
            // residual skip: layer1 output += layer1 input (= layer0 output)
            out[oidx] = y + out[(((size_t)bb_) * NT + tt) * (2 * NI) + dir * NI + j];
          }
        }
        __threadfence();
      }
      ++ep_d; gbar(bar_d, 32u * ep_d);
    } // timestep

    // ---- layer transition: reset h state, sync both directions ----
    ++ep_a; gbar(bar_a, 64u * ep_a);
    if (slice == 0) {
      for (int i = tid; i < NB * NI; i += 256) { h_prev[i] = 0.0f; h_b[i] = f2bf(0.0f); }
      __threadfence();
    }
    ++ep_a; gbar(bar_a, 64u * ep_a);
  } // layer
}

// ---------------------------------------------------------------------------
extern "C" void kernel_launch(void* const* d_in, const int* in_sizes, int n_in,
                              void* d_out, int out_size, void* d_ws, size_t ws_size,
                              hipStream_t stream) {
  (void)in_sizes; (void)n_in; (void)out_size; (void)ws_size;
  const float* x_in  = (const float*)d_in[0];
  const int*   maskp = (const int*)d_in[1];
  const float* Wx    = (const float*)d_in[2];
  const float* Wh    = (const float*)d_in[3];
  const float* bias  = (const float*)d_in[4];
  const float* Wp    = (const float*)d_in[5];
  float* out = (float*)d_out;

  // workspace carve-up (~164 MB)
  char* ws = (char*)d_ws;
  unsigned* bars  = (unsigned*)ws;                       // 256 B reserved
  float* h_acc    = (float*)(ws + 256);                  // 2*32*512 f32
  float* h_prev   = h_acc + 2 * NB * NI;                 // 2*32*512 f32
  bf16_t* h_b     = (bf16_t*)(h_prev + 2 * NB * NI);     // 2*32*512 bf16
  bf16_t* x0b     = h_b + 2 * NB * NI;                   // 32*128*512 bf16
  bf16_t* y0b     = x0b + (size_t)NB * NT * NI;          // 2*32*128*512 bf16
  bf16_t* w1      = y0b + (size_t)2 * NB * NT * NI;      // 4*16384*1024 bf16
  bf16_t* wpb     = w1 + (size_t)4 * NG * KK;            // 4*4096*512 bf16

  prep_zero<<<1, 256, 0, stream>>>(h_acc, h_prev, h_b, bars);

  const int nx = NB * NT * NI;
  prep_x<<<(nx + 255) / 256, 256, 0, stream>>>(x_in, x0b, nx);

  const size_t w1_threads = ((size_t)4 * NG * KK) / 8;   // 8,388,608
  prep_pack_w1<<<(unsigned)((w1_threads + 255) / 256), 256, 0, stream>>>(Wx, Wh, w1);

  const size_t wp_threads = ((size_t)4 * NC * NI) / 8;   // 1,048,576
  prep_pack_wp<<<(unsigned)((wp_threads + 255) / 256), 256, 0, stream>>>(Wp, wpb);

  lstm_main<<<64, 256, 0, stream>>>(maskp, bias, out, x0b, y0b, w1, wpb,
                                    h_acc, h_prev, h_b, bars);
}
